// LocalBlock_12756052869257
// MI455X (gfx1250) — matrix operand order repaired and checked by reading, not compiled, not run on hardware
//
#include <hip/hip_runtime.h>
#include <hip/hip_bf16.h>
#include <math.h>

// Problem constants (match reference)
#define Bc   4
#define Tc   2048
#define Dc   256
#define Hc   8
#define HDc  32
#define FFc  1024
#define Mc   (Bc * Tc)      // 8192 tokens
#define MAXKT 12            // max key tiles per query block (window 64 -> 9)

typedef __attribute__((ext_vector_type(16))) _Float16 v16h;
typedef __attribute__((ext_vector_type(8)))  _Float16 v8h;
typedef __attribute__((ext_vector_type(8)))  float    v8f;

static __device__ inline v8f wmma_f16(v16h a, v16h b, v8f c) {
  return __builtin_amdgcn_wmma_f32_16x16x32_f16(false, a, false, b, (short)0, c,
                                                false, false);
}

static __device__ inline v16h concat8(v8h a, v8h b) {
  return __builtin_shufflevector(a, b, 0, 1, 2, 3, 4, 5, 6, 7,
                                 8, 9, 10, 11, 12, 13, 14, 15);
}

// A-fragment (16x32 f16, row-major src, leading dim ld):
// lane m = lane&15, half = lane>>4; k = half*16 + t (one contiguous 16-half run)
static __device__ inline v16h load_A_frag(const _Float16* src, int ld, int lane) {
  const int m = lane & 15, half = lane >> 4;
  const _Float16* p = src + m * ld + half * 16;
  return concat8(*reinterpret_cast<const v8h*>(p),
                 *reinterpret_cast<const v8h*>(p + 8));
}

// B-fragment from TRANSPOSED tile srcT[n][k] (ld = k-stride):
// lane n = lane&15, half = lane>>4; k = half*16 + t -> one contiguous 16-half run
static __device__ inline v16h load_B_fragT(const _Float16* srcT, int ld, int lane) {
  const int n = lane & 15, half = lane >> 4;
  return *reinterpret_cast<const v16h*>(srcT + n * ld + half * 16);
}

// CDNA5 async copy: per-lane 16B global -> LDS (ASYNCcnt-tracked)
static __device__ inline void async_load_b128(unsigned lds_off, const void* gptr) {
  asm volatile("global_load_async_to_lds_b128 %0, %1, off"
               :: "v"(lds_off), "v"(gptr) : "memory");
}
static __device__ inline void wait_async0() {
  asm volatile("s_wait_asynccnt 0" ::: "memory");
}
// In-order async loads: <=3 outstanding means the previous tile's 3 are done.
static __device__ inline void wait_async3() {
  asm volatile("s_wait_asynccnt 3" ::: "memory");
}
static __device__ inline unsigned lds_off32(const void* p) {
  // generic LDS pointer carries the LDS byte offset in its low 32 bits
  return (unsigned)(unsigned long long)p;
}

// ------------------- f32 -> f16 weight convert + transpose: dst[n*K+k] = src[k*N+n]
__global__ __launch_bounds__(256) void cvt_t_kernel(const float* __restrict__ src,
                                                    _Float16* __restrict__ dst,
                                                    int K, int N) {
  int idx = blockIdx.x * blockDim.x + threadIdx.x;
  if (idx < K * N) {
    int n = idx / K, k = idx - n * K;
    dst[idx] = (_Float16)src[k * N + n];
  }
}

// ------------------- LayerNorm (f32 in, f16 out), wave per row -------------------
__global__ __launch_bounds__(128) void ln_f16_kernel(const float* __restrict__ x,
                                                     const float* __restrict__ g,
                                                     const float* __restrict__ bta,
                                                     _Float16* __restrict__ out, int rows) {
  const int lane = threadIdx.x & 31, wave = threadIdx.x >> 5;
  const int row = blockIdx.x * 4 + wave;
  if (row >= rows) return;
  const float* xr = x + (size_t)row * Dc;
  float v[8];
  float s = 0.f;
#pragma unroll
  for (int i = 0; i < 8; ++i) { v[i] = xr[lane * 8 + i]; s += v[i]; }
#pragma unroll
  for (int off = 1; off < 32; off <<= 1) s += __shfl_xor(s, off, 32);
  const float mean = s * (1.0f / Dc);
  float var = 0.f;
#pragma unroll
  for (int i = 0; i < 8; ++i) { float d = v[i] - mean; var += d * d; }
#pragma unroll
  for (int off = 1; off < 32; off <<= 1) var += __shfl_xor(var, off, 32);
  const float inv = rsqrtf(var * (1.0f / Dc) + 1e-5f);
  _Float16* orow = out + (size_t)row * Dc;
#pragma unroll
  for (int i = 0; i < 8; ++i) {
    const int c = lane * 8 + i;
    orow[c] = (_Float16)((v[i] - mean) * inv * g[c] + bta[c]);
  }
}

// =================== GEMM core: 128(M) x 64(N) block, 8 waves ===================
// A row-major [M,K] f16, WT transposed weights [N,K] f16.
// Double-buffered LDS with async-to-LDS staging overlapped against WMMA.
#define GEMM_ISSUE(BUF, KS)                                                          \
  async_load_b128(a_lds0[BUF], &A[(size_t)(bm + ar) * Kd + (KS) + ac]);              \
  async_load_b128(a_lds1[BUF], &A[(size_t)(bm + ar) * Kd + (KS) + ac + 8]);          \
  async_load_b128(b_lds[BUF], &WT[(size_t)(bn + br) * Kd + (KS) + bcc]);

#define GEMM_STEP(BUF, KS)                                                           \
  {                                                                                  \
    const int ksn = (KS) + 32;                                                       \
    if (ksn < Kd) { GEMM_ISSUE((BUF) ^ 1, ksn) wait_async3(); }                      \
    else          { wait_async0(); }                                                 \
    __syncthreads();                                                                 \
    v16h af = load_A_frag(&As[BUF][wm * 32], 32, lane);                              \
    _Pragma("unroll")                                                                \
    for (int tl = 0; tl < 4; ++tl) {                                                 \
      v16h bf = load_B_fragT(&BsT[BUF][tl * 16 * 32], 32, lane);                     \
      acc[tl] = wmma_f16(af, bf, acc[tl]);                                           \
    }                                                                                \
    __syncthreads();  /* frees this buffer for the next step's issue */          \
  }

// Variadic so the epilogue block may contain top-level commas.
#define GEMM_BODY(...)                                                               \
  __shared__ __attribute__((aligned(64))) _Float16 As[2][128 * 32];                  \
  __shared__ __attribute__((aligned(64))) _Float16 BsT[2][64 * 32];                  \
  const int tid = threadIdx.x, lane = tid & 31, wave = tid >> 5;                     \
  const int bm = blockIdx.x * 128, bn = blockIdx.y * 64;                             \
  const int wm = wave * 16;                                                          \
  const int ar = tid >> 1, ac = (tid & 1) * 16;   /* A: 128 rows x 32, 32B/thread */ \
  const int br = tid >> 2, bcc = (tid & 3) * 8;   /* B: 64 rows x 32, 16B/thread */  \
  const unsigned a_lds0[2] = { lds_off32(&As[0][ar * 32 + ac]),                      \
                               lds_off32(&As[1][ar * 32 + ac]) };                    \
  const unsigned a_lds1[2] = { lds_off32(&As[0][ar * 32 + ac + 8]),                  \
                               lds_off32(&As[1][ar * 32 + ac + 8]) };                \
  const unsigned b_lds[2]  = { lds_off32(&BsT[0][br * 32 + bcc]),                    \
                               lds_off32(&BsT[1][br * 32 + bcc]) };                  \
  v8f acc[4] = {};                                                                   \
  GEMM_ISSUE(0, 0)                                                                   \
  for (int ks = 0; ks < Kd; ks += 64) {   /* Kd is a multiple of 64 */               \
    GEMM_STEP(0, ks)                                                                 \
    GEMM_STEP(1, ks + 32)                                                            \
  }                                                                                  \
  const int nI = lane & 15, half = lane >> 4;                                        \
  _Pragma("unroll")                                                                  \
  for (int r = 0; r < 8; ++r) {                                                      \
    const int m = bm + wm + half * 8 + r;                                            \
    _Pragma("unroll")                                                                \
    for (int tl = 0; tl < 4; ++tl) {                                                 \
      const int n = bn + tl * 16 + nI;                                               \
      const float val = acc[tl][r] + bias[n];                                        \
      __VA_ARGS__;                                                                   \
    }                                                                                \
  }

// ------------------- QKV GEMM: [M,256] x [256,768]^T, scatter to heads -----------
__global__ __launch_bounds__(256) void gemm_qkv_kernel(
    const _Float16* __restrict__ A, const _Float16* __restrict__ WT,
    const float* __restrict__ bias, _Float16* __restrict__ Qh,
    _Float16* __restrict__ Kh, _Float16* __restrict__ Vth) {
  const int Kd = Dc;
  GEMM_BODY({
    const int b = m >> 11;
    const int t = m & (Tc - 1);
    const int which = n >> 8;
    const int cc = n & 255;
    const int h = cc >> 5;
    const int d = cc & 31;
    const size_t bh = (size_t)b * Hc + h;
    const _Float16 hv = (_Float16)val;
    if (which == 0)      Qh[(bh * Tc + t) * HDc + d] = hv;
    else if (which == 1) Kh[(bh * Tc + t) * HDc + d] = hv;
    else                 Vth[(bh * HDc + d) * Tc + t] = hv;  // V transposed [B,H,HD,T]
  })
}

// ------------------- Generic GEMM, modes: 0 = +bias+resid -> f32 ; 1 = gelu -> f16
__global__ __launch_bounds__(256) void gemm_gen_kernel(
    const _Float16* __restrict__ A, const _Float16* __restrict__ WT,
    const float* __restrict__ bias, int Nd, int Kd, int mode,
    const float* __restrict__ resid, float* __restrict__ outf,
    _Float16* __restrict__ outh) {
  GEMM_BODY({
    const size_t i0 = (size_t)m * Nd + n;
    if (mode == 0) outf[i0] = resid[i0] + val;
    else           outh[i0] = (_Float16)(0.5f * val * (1.0f + erff(val * 0.70710678118f)));
  })
}

// ------------------- Banded attention: wave per (b,h, 16-query tile) -------------------
__global__ __launch_bounds__(256) void attn_kernel(
    const _Float16* __restrict__ Qh, const _Float16* __restrict__ Kh,
    const _Float16* __restrict__ Vth, _Float16* __restrict__ Oh,
    const int* __restrict__ wptr) {
  __shared__ __attribute__((aligned(64))) _Float16 Pbuf[8][16 * 32];
  const int lane = threadIdx.x & 31, wave = threadIdx.x >> 5;
  const int task = blockIdx.x * 8 + wave;
  const int qb = task & 127, bh = task >> 7;          // T/16 == 128
  const int b = bh >> 3, h = bh & 7;
  const int q0 = qb * 16;
  const int Wn = *wptr;
  const int nI = lane & 15, half = lane >> 4;
  const float scale = 0.17677669529663687f;           // 1/sqrt(32)

  // Q fragment (A-layout): two contiguous 16B runs per lane
  const _Float16* qbase = Qh + ((size_t)bh * Tc + q0 + nI) * HDc + half * 16;
  v16h qf = concat8(*reinterpret_cast<const v8h*>(qbase),
                    *reinterpret_cast<const v8h*>(qbase + 8));

  int lo = q0 - Wn;       if (lo < 0) lo = 0;
  int hi = q0 + 15 + Wn;  if (hi > Tc - 1) hi = Tc - 1;
  const int kt0 = lo >> 4, kt1 = hi >> 4;
  int nkt = kt1 - kt0 + 1;
  if (nkt > MAXKT) nkt = MAXKT;

  float sc[MAXKT][8];
#pragma unroll
  for (int i = 0; i < MAXKT; ++i)
#pragma unroll
    for (int r = 0; r < 8; ++r) sc[i][r] = -1e30f;

#pragma unroll
  for (int i = 0; i < MAXKT; ++i) {
    if (i < nkt) {                                   // wave-uniform branch
      const int k0 = (kt0 + i) * 16;
      // B-fragment: one contiguous 32B run per lane ([B,H,T,HD] layout)
      v16h bf = *reinterpret_cast<const v16h*>(
          Kh + ((size_t)bh * Tc + k0 + nI) * HDc + half * 16);
      v8f s = {};
      s = wmma_f16(qf, bf, s);
      const int j = k0 + nI;
#pragma unroll
      for (int r = 0; r < 8; ++r) {
        const int qrow = q0 + half * 8 + r;
        float val = s[r] * scale;
        if (j - qrow > Wn || qrow - j > Wn) val = -1e30f;
        sc[i][r] = val;
      }
    }
  }

  // Row softmax (row values live across the 16 lanes of each half-group)
  float rinv[8];
#pragma unroll
  for (int r = 0; r < 8; ++r) {
    float mx = -1e30f;
#pragma unroll
    for (int i = 0; i < MAXKT; ++i) mx = fmaxf(mx, sc[i][r]);
#pragma unroll
    for (int off = 1; off < 16; off <<= 1) mx = fmaxf(mx, __shfl_xor(mx, off, 32));
    float sm = 0.f;
#pragma unroll
    for (int i = 0; i < MAXKT; ++i) { float p = __expf(sc[i][r] - mx); sc[i][r] = p; sm += p; }
#pragma unroll
    for (int off = 1; off < 16; off <<= 1) sm += __shfl_xor(sm, off, 32);
    rinv[r] = 1.0f / sm;
  }

  // P @ V over key-tile pairs (K = 32 per WMMA)
  v8f o0 = {}, o1 = {};
  const int npair = (nkt + 1) >> 1;
  _Float16* Pb = &Pbuf[wave][0];
#pragma unroll
  for (int pp = 0; pp < MAXKT / 2; ++pp) {
    if (pp < npair) {                                // wave-uniform branch
#pragma unroll
      for (int r = 0; r < 8; ++r) {
        const int m = half * 8 + r;
#pragma unroll
        for (int tl = 0; tl < 2; ++tl)
          Pb[m * 32 + tl * 16 + nI] = (_Float16)(sc[2 * pp + tl][r] * rinv[r]);
      }
      asm volatile("s_wait_dscnt 0" ::: "memory");   // intra-wave LDS store->gather
      v16h pf = load_A_frag(Pb, 32, lane);
      const int pk0 = (kt0 + 2 * pp) * 16;
      int kb = pk0 + half * 16;                      // tile-level clamp keeps load vector
      if (kb > Tc - 16) kb = Tc - 16;                // (clamped rows hit zero probs)
#pragma unroll
      for (int dh = 0; dh < 2; ++dh) {
        v16h vf = *reinterpret_cast<const v16h*>(
            Vth + ((size_t)bh * HDc + dh * 16 + nI) * Tc + kb);
        if (dh == 0) o0 = wmma_f16(pf, vf, o0);
        else         o1 = wmma_f16(pf, vf, o1);
      }
    }
  }

  // Store O as [M, D] f16 for the out-projection GEMM
#pragma unroll
  for (int r = 0; r < 8; ++r) {
    const int tq = q0 + half * 8 + r;
    const size_t base = ((size_t)(b * Tc + tq)) * Dc + h * HDc;
    Oh[base + nI]      = (_Float16)o0[r];
    Oh[base + 16 + nI] = (_Float16)o1[r];
  }
}

// ------------------------------- host launcher -------------------------------
extern "C" void kernel_launch(void* const* d_in, const int* in_sizes, int n_in,
                              void* d_out, int out_size, void* d_ws, size_t ws_size,
                              hipStream_t stream) {
  const float* x        = (const float*)d_in[0];
  const float* ln1_g    = (const float*)d_in[1];
  const float* ln1_b    = (const float*)d_in[2];
  const float* in_projw = (const float*)d_in[3];
  const float* in_projb = (const float*)d_in[4];
  const float* out_w    = (const float*)d_in[5];
  const float* out_b    = (const float*)d_in[6];
  const float* ln2_g    = (const float*)d_in[7];
  const float* ln2_b    = (const float*)d_in[8];
  const float* w1       = (const float*)d_in[9];
  const float* b1       = (const float*)d_in[10];
  const float* w2       = (const float*)d_in[11];
  const float* b2       = (const float*)d_in[12];
  const int*   window   = (const int*)d_in[13];
  float* out = (float*)d_out;

  char* ws = (char*)d_ws;
  size_t off = 0;
  auto alloc = [&](size_t elems, size_t esz) {
    void* p = ws + off;
    off += (elems * esz + 255) & ~(size_t)255;
    return p;
  };
  // Transposed f16 weights: [N][K]
  _Float16* wqkv_t = (_Float16*)alloc((size_t)3 * Dc * Dc, 2);
  _Float16* wo_t   = (_Float16*)alloc((size_t)Dc * Dc, 2);
  _Float16* w1_t   = (_Float16*)alloc((size_t)FFc * Dc, 2);
  _Float16* w2_t   = (_Float16*)alloc((size_t)Dc * FFc, 2);
  _Float16* xn_h   = (_Float16*)alloc((size_t)Mc * Dc, 2);
  _Float16* Qh     = (_Float16*)alloc((size_t)Mc * Dc, 2);
  _Float16* Kh     = (_Float16*)alloc((size_t)Mc * Dc, 2);
  _Float16* Vth    = (_Float16*)alloc((size_t)Mc * Dc, 2);
  _Float16* Oh     = (_Float16*)alloc((size_t)Mc * Dc, 2);
  float*    y      = (float*)   alloc((size_t)Mc * Dc, 4);
  _Float16* ffin_h = (_Float16*)alloc((size_t)Mc * Dc, 2);
  _Float16* g_h    = (_Float16*)alloc((size_t)Mc * FFc, 2);

  // 1) weights -> f16, transposed to [N][K]
  cvt_t_kernel<<<(3 * Dc * Dc + 255) / 256, 256, 0, stream>>>(in_projw, wqkv_t, Dc, 3 * Dc);
  cvt_t_kernel<<<(Dc * Dc + 255) / 256, 256, 0, stream>>>(out_w, wo_t, Dc, Dc);
  cvt_t_kernel<<<(Dc * FFc + 255) / 256, 256, 0, stream>>>(w1, w1_t, Dc, FFc);
  cvt_t_kernel<<<(FFc * Dc + 255) / 256, 256, 0, stream>>>(w2, w2_t, FFc, Dc);

  // 2) LN1
  ln_f16_kernel<<<Mc / 4, 128, 0, stream>>>(x, ln1_g, ln1_b, xn_h, Mc);

  // 3) QKV projection + head scatter
  gemm_qkv_kernel<<<dim3(Mc / 128, (3 * Dc) / 64), 256, 0, stream>>>(
      xn_h, wqkv_t, in_projb, Qh, Kh, Vth);

  // 4) banded attention
  attn_kernel<<<(Bc * Hc * (Tc / 16)) / 8, 256, 0, stream>>>(Qh, Kh, Vth, Oh, window);

  // 5) out projection + residual (y = x + attn_out)
  gemm_gen_kernel<<<dim3(Mc / 128, Dc / 64), 256, 0, stream>>>(
      Oh, wo_t, out_b, Dc, Dc, 0, x, y, ((_Float16*)nullptr));

  // 6) LN2
  ln_f16_kernel<<<Mc / 4, 128, 0, stream>>>(y, ln2_g, ln2_b, ffin_h, Mc);

  // 7) FF1 + exact GELU
  gemm_gen_kernel<<<dim3(Mc / 128, FFc / 64), 256, 0, stream>>>(
      ffin_h, w1_t, b1, FFc, Dc, 1, (const float*)nullptr, (float*)nullptr, g_h);

  // 8) FF2 + residual from x -> out
  gemm_gen_kernel<<<dim3(Mc / 128, Dc / 64), 256, 0, stream>>>(
      g_h, w2_t, b2, Dc, FFc, 0, x, out, ((_Float16*)nullptr));
}
